// SDPA_54743653155297
// MI455X (gfx1250) — matrix operand order repaired and checked by reading, not compiled, not run on hardware
//
#include <hip/hip_runtime.h>
#include <hip/hip_bf16.h>

// ---------------------------------------------------------------------------
// CDNA5 (gfx1250) fused QKV + SDPA, bf16 WMMA path + async global->LDS.
// ---------------------------------------------------------------------------

typedef __attribute__((ext_vector_type(16))) __bf16 v16bf;
typedef __attribute__((ext_vector_type(8)))  float  v8f;

__device__ __forceinline__ unsigned short f2bf_raw(float f) {
  union { float f; unsigned u; } x; x.f = f;
  unsigned r = x.u + 0x7fffu + ((x.u >> 16) & 1u);   // round-to-nearest-even
  return (unsigned short)(r >> 16);
}

// Build a v16bf fragment from two 16-byte LDS chunks (two ds_load_b128).
__device__ __forceinline__ v16bf frag_from(const unsigned short* lds, int off0, int off1) {
  union { uint4 u[2]; v16bf v; } x;
  x.u[0] = *reinterpret_cast<const uint4*>(lds + off0);
  x.u[1] = *reinterpret_cast<const uint4*>(lds + off1);
  return x.v;
}

__device__ __forceinline__ v8f wmma_bf16(v16bf a, v16bf b, v8f c) {
  // (neg_a, A, neg_b, B, c_mod, C, reuse_a, reuse_b)
  return __builtin_amdgcn_wmma_f32_16x16x32_bf16(false, a, false, b, (short)0, c,
                                                 false, false);
}

// gfx1250 async global->LDS copy, 16B per lane, tracked by ASYNCcnt.
// VDST VGPR = wave-relative LDS byte address (low 32 bits of the flat shared
// pointer), ADDR = 64-bit global address (GV mode).
__device__ __forceinline__ void async_ld_b128(unsigned lds_addr, const void* gaddr) {
  asm volatile("global_load_async_to_lds_b128 %0, %1, off"
               :: "v"(lds_addr), "v"(gaddr)
               : "memory");
}
__device__ __forceinline__ void s_wait_async0() {
  asm volatile("s_wait_asynccnt 0x0" ::: "memory");
}

// ---------------------------------------------------------------------------
// Generic tiled GEMM: C[z] = alpha * A[z] (*) op(B[z]) + bias
//   BM=64, BN=128, KT=32; 256 threads = 8 waves; each wave -> 32x32 of C.
//   A_BF16 / B_BF16 : source element type (fp32 sources converted to bf16
//                      while staging to LDS; bf16 A / NT-path bf16 B staged with
//                      global_load_async_to_lds_b128).
//   B_TRANS         : B stored row-major [N,K] (NT gemm, contiguous K reads).
//   OUT_BF16        : C written as bf16 (raw ushort), else fp32.
//   All of M % 64, N % 128, K % 32 are zero for our launches.
// ---------------------------------------------------------------------------
template <bool A_BF16, bool B_BF16, bool B_TRANS, bool BIAS, bool OUT_BF16>
__global__ __launch_bounds__(256) void gemm_wmma(
    const void* __restrict__ Ap, const void* __restrict__ Bp,
    const float* __restrict__ bias, void* __restrict__ Cp,
    int M, int N, int K, int lda, int ldb, int ldc, float alpha,
    long long sA, long long sB, long long sC)
{
  constexpr int BM = 64, BN = 128, KT = 32;
  constexpr bool ASYNC_A = A_BF16;
  constexpr bool ASYNC_B = (B_TRANS && B_BF16);
  __shared__ alignas(16) unsigned short As[BM * KT];   // [row][k]   4 KB
  __shared__ alignas(16) unsigned short Bt[BN * KT];   // [col][k]   8 KB

  const int t    = threadIdx.x;
  const int wave = t >> 5;
  const int lane = t & 31;
  const int wm   = wave & 1;    // 2 waves along M  (2*32 = 64)
  const int wn   = wave >> 1;   // 4 waves along N  (4*32 = 128)
  const int m0   = blockIdx.y * BM;
  const int n0   = blockIdx.x * BN;
  const long long zb = blockIdx.z;

  const float*          A32 = (const float*)Ap;
  const unsigned short* A16 = (const unsigned short*)Ap;
  const float*          B32 = (const float*)Bp;
  const unsigned short* B16 = (const unsigned short*)Bp;
  if constexpr (A_BF16) A16 += zb * sA; else A32 += zb * sA;
  if constexpr (B_BF16) B16 += zb * sB; else B32 += zb * sB;
  float*          C32 = (float*)Cp + zb * sC;
  unsigned short* C16 = (unsigned short*)Cp + zb * sC;

  v8f acc00 = {}, acc01 = {}, acc10 = {}, acc11 = {};

  // staging coordinates (fixed per thread)
  const int a_row = t >> 2;            // 0..63
  const int a_kb  = (t & 3) * 8;       // 8 elements each
  const int bt_n  = t >> 1;            // 0..127   (NT path: 16 elements each)
  const int bt_kb = (t & 1) * 16;
  const int bn_k  = t >> 3;            // 0..31    (NN path)
  const int bn_nb = (t & 7) * 16;

  const int half = lane >> 4;          // 0 | 1  (lane group)
  const int lrow = lane & 15;
  const int akb  = half * 8;           // A frag: K 0..7/16..23 vs 8..15/24..31
  const int bkb  = half * 8;           // B frag: K 0..7/16..23 vs 8..15/24..31

  for (int k0 = 0; k0 < K; k0 += KT) {
    // ---- stage A tile (64 x 32) as bf16 --------------------------------
    {
      const long long g = (long long)(m0 + a_row) * lda + k0 + a_kb;
      if constexpr (ASYNC_A) {
        async_ld_b128((unsigned)(size_t)&As[a_row * KT + a_kb],
                      (const void*)(A16 + g));
      } else {
        float4 f0 = *reinterpret_cast<const float4*>(A32 + g);
        float4 f1 = *reinterpret_cast<const float4*>(A32 + g + 4);
        union { unsigned short s[8]; uint4 u; } pk;
        pk.s[0] = f2bf_raw(f0.x); pk.s[1] = f2bf_raw(f0.y);
        pk.s[2] = f2bf_raw(f0.z); pk.s[3] = f2bf_raw(f0.w);
        pk.s[4] = f2bf_raw(f1.x); pk.s[5] = f2bf_raw(f1.y);
        pk.s[6] = f2bf_raw(f1.z); pk.s[7] = f2bf_raw(f1.w);
        *reinterpret_cast<uint4*>(&As[a_row * KT + a_kb]) = pk.u;
        if (k0 + KT < K) __builtin_prefetch(A32 + g + KT, 0, 1);
      }
    }
    // ---- stage B tile as Bt[col][k] (128 x 32) bf16 --------------------
    if constexpr (B_TRANS) {   // B row-major [N,K]: contiguous copy
      const long long g = (long long)(n0 + bt_n) * ldb + k0 + bt_kb;
      if constexpr (B_BF16) {
        async_ld_b128((unsigned)(size_t)&Bt[bt_n * KT + bt_kb],
                      (const void*)(B16 + g));
        async_ld_b128((unsigned)(size_t)&Bt[bt_n * KT + bt_kb + 8],
                      (const void*)(B16 + g + 8));
      } else {
        for (int i = 0; i < 16; ++i)
          Bt[bt_n * KT + bt_kb + i] = f2bf_raw(B32[g + i]);
      }
    } else {                   // B row-major [K,N]: transpose while staging
      const long long g = (long long)(k0 + bn_k) * ldb + n0 + bn_nb;
      if constexpr (B_BF16) {
        union { uint4 u[2]; unsigned short s[16]; } tb;
        tb.u[0] = *reinterpret_cast<const uint4*>(B16 + g);
        tb.u[1] = *reinterpret_cast<const uint4*>(B16 + g + 8);
        for (int i = 0; i < 16; ++i) Bt[(bn_nb + i) * KT + bn_k] = tb.s[i];
        if (k0 + KT < K) __builtin_prefetch(B16 + g + (long long)KT * ldb, 0, 1);
      } else {
        for (int i = 0; i < 16; ++i)
          Bt[(bn_nb + i) * KT + bn_k] = f2bf_raw(B32[g + i]);
        if (k0 + KT < K) __builtin_prefetch(B32 + g + (long long)KT * ldb, 0, 1);
      }
    }
    if constexpr (ASYNC_A || ASYNC_B) s_wait_async0();   // own async copies done
    __syncthreads();                                     // whole tile staged

    // ---- fragments + 4x v_wmma_f32_16x16x32_bf16 -----------------------
    const int ar0 = (wm * 32 +  0 + lrow) * KT + akb;
    const int ar1 = (wm * 32 + 16 + lrow) * KT + akb;
    v16bf a0 = frag_from(As, ar0, ar0 + 16);
    v16bf a1 = frag_from(As, ar1, ar1 + 16);
    const int bc0 = (wn * 32 +  0 + lrow) * KT + bkb;
    const int bc1 = (wn * 32 + 16 + lrow) * KT + bkb;
    v16bf b0 = frag_from(Bt, bc0, bc0 + 16);
    v16bf b1 = frag_from(Bt, bc1, bc1 + 16);

    acc00 = wmma_bf16(a0, b0, acc00);
    acc01 = wmma_bf16(a0, b1, acc01);
    acc10 = wmma_bf16(a1, b0, acc10);
    acc11 = wmma_bf16(a1, b1, acc11);
    __syncthreads();                                     // readers done (WAR)
  }

  // ---- epilogue: C/D layout = VGPR r -> M = r + (lane>=16 ? 8 : 0) -----
  const int ccol = lane & 15;
  const int radd = (lane >> 4) * 8;
  auto store_tile = [&](const v8f& acc, int tm, int tn) {
    const int bm = m0 + wm * 32 + tm * 16 + radd;
    const int bn = n0 + wn * 32 + tn * 16 + ccol;
    float bv = 0.0f;
    if constexpr (BIAS) bv = bias[bn];
    for (int r = 0; r < 8; ++r) {
      const float v = acc[r] * alpha + bv;
      const long long off = (long long)(bm + r) * ldc + bn;
      if constexpr (OUT_BF16) C16[off] = f2bf_raw(v);
      else                    C32[off] = v;
    }
  };
  store_tile(acc00, 0, 0); store_tile(acc01, 0, 1);
  store_tile(acc10, 1, 0); store_tile(acc11, 1, 1);
}

// ---------------------------------------------------------------------------
// Row softmax (in place), one 256-thread block per row of 4096 fp32.
// ---------------------------------------------------------------------------
__global__ __launch_bounds__(256) void softmax_rows(float* __restrict__ attn, int ncols)
{
  float* p = attn + (long long)blockIdx.x * ncols;
  const int t = threadIdx.x;
  __shared__ float red[256];

  float m = -3.4e38f;
  for (int c = t; c < ncols; c += 256) m = fmaxf(m, p[c]);
  red[t] = m; __syncthreads();
  for (int s = 128; s > 0; s >>= 1) {
    if (t < s) red[t] = fmaxf(red[t], red[t + s]);
    __syncthreads();
  }
  m = red[0]; __syncthreads();

  float sum = 0.0f;
  for (int c = t; c < ncols; c += 256) {
    float e = __expf(p[c] - m);
    p[c] = e;
    sum += e;
  }
  red[t] = sum; __syncthreads();
  for (int s = 128; s > 0; s >>= 1) {
    if (t < s) red[t] += red[t + s];
    __syncthreads();
  }
  const float inv = 1.0f / red[0];
  for (int c = t; c < ncols; c += 256) p[c] *= inv;
}

// ---------------------------------------------------------------------------
// Launch: x[4,4096,512] fp32, W_qkv[512,1536] fp32, b_qkv[1536] fp32
// d_out = opt[4,4096,512] fp32 ++ attn[4,4096,4096] fp32
// d_ws  = qkv bf16 [16384, 1536]  (48 MB)
// ---------------------------------------------------------------------------
extern "C" void kernel_launch(void* const* d_in, const int* in_sizes, int n_in,
                              void* d_out, int out_size, void* d_ws, size_t ws_size,
                              hipStream_t stream)
{
  const float* x = (const float*)d_in[0];
  const float* W = (const float*)d_in[1];
  const float* b = (const float*)d_in[2];

  const int B = 4, S = 4096, H = 512;
  const int M  = B * S;       // 16384 rows of x
  const int N3 = 3 * H;       // 1536

  float* opt  = (float*)d_out;
  float* attn = (float*)d_out + (long long)B * S * H;
  unsigned short* qkv = (unsigned short*)d_ws;     // bf16 [M, 1536]

  const dim3 blk(256);

  // 1) qkv = x @ W + b     (fp32 x fp32 -> bf16, NN, +bias)
  gemm_wmma<false, false, false, true, true>
      <<<dim3(N3 / 128, M / 64, 1), blk, 0, stream>>>(
          x, W, b, qkv, M, N3, H, H, N3, N3, 1.0f, 0, 0, 0);

  // 2) scores = (q @ k^T) * 1/sqrt(H)   (bf16 x bf16 -> fp32, NT, async)
  const float scale = 0.044194173824159216f;   // 1/sqrt(512)
  gemm_wmma<true, true, true, false, false>
      <<<dim3(S / 128, S / 64, B), blk, 0, stream>>>(
          qkv, qkv + H, nullptr, attn, S, S, H, N3, N3, S, scale,
          (long long)S * N3, (long long)S * N3, (long long)S * S);

  // 3) softmax over rows of attn
  softmax_rows<<<dim3(B * S), blk, 0, stream>>>(attn, S);

  // 4) opt = attn @ v      (fp32 x bf16 -> fp32, NN)
  gemm_wmma<false, true, false, false, false>
      <<<dim3(H / 128, S / 64, B), blk, 0, stream>>>(
          attn, qkv + 2 * H, nullptr, opt, S, H, S, S, N3, H, 1.0f,
          (long long)S * S, (long long)S * N3, (long long)S * H);
}
